// GNNModel_64381559767781
// MI455X (gfx1250) — matrix operand order repaired and checked by reading, not compiled, not run on hardware
//
#include <hip/hip_runtime.h>

// GNN: 2x GraphConv(D=16) + MLP head, fp32 data path, WMMA f16 (f32 acc) for
// the per-node 16x16 transforms. Bandwidth-bound; agg/x arrays (32MB) are
// L2-resident on MI455X (192MB L2).

typedef __attribute__((ext_vector_type(16))) _Float16 v16h;
typedef __attribute__((ext_vector_type(8)))  float    v8f;

#define D 16

// ---------------------------------------------------------------- zero
__global__ void __launch_bounds__(256)
zero_kernel(float* __restrict__ p, int n4) {
  int i = blockIdx.x * blockDim.x + threadIdx.x;
  if (i < n4) ((float4*)p)[i] = make_float4(0.f, 0.f, 0.f, 0.f);
}

// ------------------------------------------------- edge gather + scatter-add
// 4 threads per edge, float4 per thread: one 64B coalesced row gather per
// edge, 16x global_atomic_add_f32 into agg[dst].
__global__ void __launch_bounds__(256)
scatter_kernel(const float* __restrict__ x,
               const int* __restrict__ src,
               const int* __restrict__ dst,
               float* __restrict__ agg,
               int n_edges) {
  long long gid = (long long)blockIdx.x * blockDim.x + threadIdx.x;
  int e    = (int)(gid >> 2);
  int part = (int)(gid & 3);
  if (e >= n_edges) return;
  int s = src[e];
  int d = dst[e];
  const float4 v = *(const float4*)(x + (size_t)s * D + part * 4);
  float* o = agg + (size_t)d * D + part * 4;
  atomicAdd(o + 0, v.x);
  atomicAdd(o + 1, v.y);
  atomicAdd(o + 2, v.z);
  atomicAdd(o + 3, v.w);
}

// ------------------------------------------- per-node transform via WMMA
// One wave per 16-node tile. Single v_wmma_f32_16x16x32_f16 computes
//   OUT(16x16) = [AGG | X](16x32) @ [Wrel^T ; Wroot^T](32x16)
// i.e. agg@Wrel^T + x@Wroot^T, then +bias, ReLU (always, both conv layers).
// NOTE: `agg` may alias `out` (layer 2) -> no __restrict__ on them.
__global__ void __launch_bounds__(256)
node_transform_wmma(const float* agg,
                    const float* __restrict__ xin,
                    const float* __restrict__ w_rel,   // [16,16] row-major
                    const float* __restrict__ w_root,  // [16,16] row-major
                    const float* __restrict__ b_rel,   // [16]
                    float* out,
                    int num_tiles, int n_nodes) {
  int wave = threadIdx.x >> 5;
  int tile = blockIdx.x * (blockDim.x >> 5) + wave;
  if (tile >= num_tiles) return;           // wave-uniform: EXEC stays all-1s
  int lane = threadIdx.x & 31;
  int r    = lane & 15;                    // A row / B column / bias index
  int hi   = lane >> 4;

  // A fragment (16-bit 16x32): lanes 0-15 K in {0..7}u{16..23}, lanes 16-31
  // K in {8..15}u{24..31}. K<16 = AGG features, K>=16 = X features.
  int arow = tile * 16 + r;
  if (arow >= n_nodes) arow = n_nodes - 1;     // tail clamp (reads only)
  const float* aRow = agg + (size_t)arow * D + hi * 8;
  const float* xRow = xin + (size_t)arow * D + hi * 8;
  v16h a;
#pragma unroll
  for (int i = 0; i < 8; ++i) a[i] = (_Float16)aRow[i];
#pragma unroll
  for (int i = 0; i < 8; ++i) a[8 + i] = (_Float16)xRow[i];

  // B fragment (32x16): lane holds col n=r, K same as A: {hi*8..+7}u{16+hi*8..+7}.
  // B[k][n] = w_rel[n][k] (k<16) else w_root[n][k-16] -> contiguous row loads.
  const float* wrel_row  = w_rel  + r * D + hi * 8;
  const float* wroot_row = w_root + r * D + hi * 8;
  v16h b;
#pragma unroll
  for (int i = 0; i < 8; ++i) b[i] = (_Float16)wrel_row[i];
#pragma unroll
  for (int i = 0; i < 8; ++i) b[8 + i] = (_Float16)wroot_row[i];

  v8f c = {};
  c = __builtin_amdgcn_wmma_f32_16x16x32_f16(
      /*neg_a=*/false, a, /*neg_b=*/false, b,
      /*c_mod=*/(short)0, c, /*reuse_a=*/false, /*reuse_b=*/false);

  // C/D layout: VGPR v -> row v + 8*hi, col = r. Coalesced 64B stores.
  // Tile-uniform fast path: full tiles (all tiles when 16 | n_nodes) take
  // 8 unguarded stores off one base pointer -- no per-element exec toggles.
  float bias = b_rel[r];
  int rowbase = tile * 16 + hi * 8;
  float* obase = out + (size_t)rowbase * D + r;
  if (tile * 16 + 16 <= n_nodes) {         // uniform across the wave
#pragma unroll
    for (int v = 0; v < 8; ++v) {
      obase[(size_t)v * D] = fmaxf(c[v] + bias, 0.f);
    }
  } else {                                  // tail tile (rare / absent here)
#pragma unroll
    for (int v = 0; v < 8; ++v) {
      if (rowbase + v < n_nodes)
        obase[(size_t)v * D] = fmaxf(c[v] + bias, 0.f);
    }
  }
}

// ---------------------------------------------------- MLP head 16->8->2
__global__ void __launch_bounds__(256)
fc_head_kernel(const float* __restrict__ h,
               const float* __restrict__ fc1_w, const float* __restrict__ fc1_b,
               const float* __restrict__ fc2_w, const float* __restrict__ fc2_b,
               float* __restrict__ out, int n_nodes) {
  __shared__ float sw1[8 * 16];
  __shared__ float sb1[8];
  __shared__ float sw2[2 * 8];
  __shared__ float sb2[2];
  int t = threadIdx.x;
  if (t < 128)       sw1[t]       = fc1_w[t];
  else if (t < 136)  sb1[t - 128] = fc1_b[t - 128];
  else if (t < 152)  sw2[t - 136] = fc2_w[t - 136];
  else if (t < 154)  sb2[t - 152] = fc2_b[t - 152];
  __syncthreads();

  int i = blockIdx.x * blockDim.x + t;
  if (i >= n_nodes) return;

  float hv[16];
  const float4* hp = (const float4*)(h + (size_t)i * D);
#pragma unroll
  for (int q = 0; q < 4; ++q) {
    float4 v = hp[q];
    hv[4 * q] = v.x; hv[4 * q + 1] = v.y; hv[4 * q + 2] = v.z; hv[4 * q + 3] = v.w;
  }

  float s[8];
#pragma unroll
  for (int o = 0; o < 8; ++o) {
    float acc = sb1[o];
#pragma unroll
    for (int k = 0; k < 16; ++k) acc = fmaf(hv[k], sw1[o * 16 + k], acc);
    s[o] = fmaxf(acc, 0.f);
  }
#pragma unroll
  for (int o = 0; o < 2; ++o) {
    float acc = sb2[o];
#pragma unroll
    for (int k = 0; k < 8; ++k) acc = fmaf(s[k], sw2[o * 8 + k], acc);
    out[(size_t)i * 2 + o] = acc;
  }
}

// ---------------------------------------------------------------- launch
extern "C" void kernel_launch(void* const* d_in, const int* in_sizes, int n_in,
                              void* d_out, int out_size, void* d_ws, size_t ws_size,
                              hipStream_t stream) {
  const float* x    = (const float*)d_in[0];
  const int*   ei   = (const int*)d_in[1];
  const float* w1r  = (const float*)d_in[2];
  const float* b1   = (const float*)d_in[3];
  const float* w1o  = (const float*)d_in[4];
  const float* w2r  = (const float*)d_in[5];
  const float* b2   = (const float*)d_in[6];
  const float* w2o  = (const float*)d_in[7];
  const float* fc1w = (const float*)d_in[8];
  const float* fc1b = (const float*)d_in[9];
  const float* fc2w = (const float*)d_in[10];
  const float* fc2b = (const float*)d_in[11];

  const int n_nodes = in_sizes[0] / D;    // 500000
  const int n_edges = in_sizes[1] / 2;    // 5000000
  const int* src = ei;                    // edge_index[0]
  const int* dst = ei + n_edges;          // edge_index[1]

  float* agg = (float*)d_ws;                      // [N,16] f32  (32 MB)
  float* h1  = agg + (size_t)n_nodes * D;         // [N,16] f32  (32 MB)

  const int nfloats   = n_nodes * D;
  const int n4        = nfloats / 4;
  const int num_tiles = (n_nodes + 15) / 16;

  dim3 blk(256);
  int zgrid = (n4 + 255) / 256;
  int sgrid = (int)(((long long)n_edges * 4 + 255) / 256);
  int tgrid = (num_tiles + 7) / 8;                // 8 waves / block
  int fgrid = (n_nodes + 255) / 256;

  // ---- layer 1: h1 = relu(agg(x)@W1rel^T + b1 + x@W1root^T)
  zero_kernel   <<<zgrid, blk, 0, stream>>>(agg, n4);
  scatter_kernel<<<sgrid, blk, 0, stream>>>(x, src, dst, agg, n_edges);
  node_transform_wmma<<<tgrid, blk, 0, stream>>>(agg, x, w1r, w1o, b1, h1,
                                                 num_tiles, n_nodes);
  // ---- layer 2: agg-buffer reused as output (safe in-tile aliasing)
  zero_kernel   <<<zgrid, blk, 0, stream>>>(agg, n4);
  scatter_kernel<<<sgrid, blk, 0, stream>>>(h1, src, dst, agg, n_edges);
  node_transform_wmma<<<tgrid, blk, 0, stream>>>(agg, h1, w2r, w2o, b2, agg,
                                                 num_tiles, n_nodes);
  // ---- head
  fc_head_kernel<<<fgrid, blk, 0, stream>>>(agg, fc1w, fc1b, fc2w, fc2b,
                                            (float*)d_out, n_nodes);
}
